// CausalMultiHeadSelfAttention_17549236372136
// MI455X (gfx1250) — matrix operand order repaired and checked by reading, not compiled, not run on hardware
//
#include <hip/hip_runtime.h>

typedef unsigned short ushort_t;
typedef __attribute__((ext_vector_type(8)))  float        v8f;
typedef __attribute__((ext_vector_type(16))) __bf16       v16bf;
typedef __attribute__((ext_vector_type(4)))  unsigned int u32x4;
typedef __attribute__((ext_vector_type(8)))  int          i32x8;
typedef __attribute__((ext_vector_type(4)))  int          i32x4;

union V16 { v16bf v; u32x4 u[2]; };

#if __has_builtin(__builtin_amdgcn_tensor_load_to_lds)
#define HAVE_TDM 1
#else
#define HAVE_TDM 0
#endif

__device__ __forceinline__ ushort_t f2bf(float f) {
  unsigned u = __builtin_bit_cast(unsigned, f);
  u += 0x7FFFu + ((u >> 16) & 1u);            // round-to-nearest-even
  return (ushort_t)(u >> 16);
}
__device__ __forceinline__ float bf2f(ushort_t h) {
  unsigned u = ((unsigned)h) << 16;
  return __builtin_bit_cast(float, u);
}

#if HAVE_TDM
// Issue one 2D Tensor-DMA load (global -> LDS). Descriptor per CDNA5 ISA ch.8.
// dims/strides in data_size (=2B) units; LDS receives tile rows packed
// contiguously (row stride = tile_d0 elements).
__device__ __forceinline__ void tdm_load_2d(unsigned lds_byte_addr,
                                            const ushort_t* gptr,
                                            unsigned tensor_d0, unsigned tensor_d1,
                                            unsigned tile_d0, unsigned tile_d1,
                                            unsigned long long stride0) {
  const unsigned long long ga = (unsigned long long)gptr;
  u32x4 g0;
  g0[0] = 1u;                                    // count=1, user descriptor
  g0[1] = lds_byte_addr;                         // D#.lds_addr
  g0[2] = (unsigned)ga;                          // global_addr[31:0]
  g0[3] = (unsigned)(ga >> 32) | (2u << 30);     // global_addr[56:32] | type=2
  i32x8 g1;
  g1[0] = (int)(1u << 16);                       // wg_mask=0 | data_size=1 (2B)
  g1[1] = (int)((tensor_d0 & 0xFFFFu) << 16);    // barrier_addr=0 | dim0[15:0]
  g1[2] = (int)((tensor_d0 >> 16) | ((tensor_d1 & 0xFFFFu) << 16));
  g1[3] = (int)((tensor_d1 >> 16) | (tile_d0 << 16));
  g1[4] = (int)(tile_d1 & 0xFFFFu);              // tile_dim1 | tile_dim2=0
  g1[5] = (int)(unsigned)(stride0 & 0xFFFFFFFFull);
  g1[6] = (int)(unsigned)((stride0 >> 32) & 0xFFFFull);  // | dim1_stride lo = 0
  g1[7] = 0;
  const i32x4 gz = (i32x4){0, 0, 0, 0};
#if defined(__clang_major__) && (__clang_major__ >= 23)
  const i32x8 gz8 = (i32x8){0, 0, 0, 0, 0, 0, 0, 0};
  __builtin_amdgcn_tensor_load_to_lds(g0, g1, gz, gz, gz8, 0);
#else
  __builtin_amdgcn_tensor_load_to_lds(g0, g1, gz, gz, 0);
#endif
}
#if __has_builtin(__builtin_amdgcn_s_wait_tensorcnt)
#define WAIT_TENSOR(n) __builtin_amdgcn_s_wait_tensorcnt((short)(n))
#else
#define WAIT_TENSOR(n) asm volatile("s_wait_tensorcnt %0" ::"i"(n) : "memory")
#endif
#endif  // HAVE_TDM

// ---------------------------------------------------------------- elementwise
__global__ void cvt_f32_to_bf16(const float* __restrict__ in,
                                ushort_t* __restrict__ out, int n) {
  int i = blockIdx.x * blockDim.x + threadIdx.x;
  if (i < n) out[i] = f2bf(in[i]);
}

// RoPE: read Q/K bf16 in [B,S,H*64], write [B,H,S,64] bf16.
__global__ void rope_kernel(const ushort_t* __restrict__ Qin,
                            const ushort_t* __restrict__ Kin,
                            const int* __restrict__ pos,
                            ushort_t* __restrict__ Qout,
                            ushort_t* __restrict__ Kout,
                            int S, int total) {
  int idx = blockIdx.x * blockDim.x + threadIdx.x;
  if (idx >= total) return;
  const int i = idx & 31;             // freq index, d_k/2 = 32
  int rest = idx >> 5;
  const int s = rest % S;  rest /= S;
  const int h = rest & 15;
  const int b = rest >> 4;

  const float p   = (float)pos[s];
  const float inv = __expf(-(float)i * (9.210340371976184f / 32.0f)); // 10000^(-i/32)
  const float ang = p * inv;
  const float c  = __cosf(ang);
  const float sn = __sinf(ang);

  const size_t inb  = ((size_t)b * S + s) * 1024 + h * 64 + 2 * i;
  const size_t outb = (((size_t)b * 16 + h) * S + s) * 64 + 2 * i;

  float q1 = bf2f(Qin[inb]), q2 = bf2f(Qin[inb + 1]);
  Qout[outb]     = f2bf(q1 * c - q2 * sn);
  Qout[outb + 1] = f2bf(q1 * sn + q2 * c);
  float k1 = bf2f(Kin[inb]), k2 = bf2f(Kin[inb + 1]);
  Kout[outb]     = f2bf(k1 * c - k2 * sn);
  Kout[outb + 1] = f2bf(k1 * sn + k2 * c);
}

// V [B,S,H*64] -> Vt [B,H,64,S]
__global__ void transpose_v(const ushort_t* __restrict__ Vin,
                            ushort_t* __restrict__ Vt, int S, int total) {
  int idx = blockIdx.x * blockDim.x + threadIdx.x;
  if (idx >= total) return;
  const int d = idx & 63;
  int rest = idx >> 6;
  const int h = rest & 15;  rest >>= 4;
  const int s = rest % S;
  const int b = rest / S;
  Vt[(((size_t)b * 16 + h) * 64 + d) * S + s] = Vin[idx];
}

// ------------------------------------------------------------------- GEMM
// C[m,n] = sum_k A[m,k] * Bw[n,k]   (A: MxK bf16 row-major, Bw: NxK bf16 row-major)
// One wave computes a 16x64 tile of C.  8 waves per block.
template <bool OUT_F32>
__global__ __launch_bounds__(256)
void gemm_bf16_nt(const ushort_t* __restrict__ A, const ushort_t* __restrict__ Bw,
                  float* __restrict__ Cf, ushort_t* __restrict__ Cb,
                  int M, int N, int K) {
  const int lane = threadIdx.x & 31;
  const int wave = threadIdx.x >> 5;
  const int nTiles = N >> 6;
  const int totalTiles = (M >> 4) * nTiles;
  const int tile = blockIdx.x * 8 + wave;
  if (tile >= totalTiles) return;
  const int m0 = (tile / nTiles) << 4;
  const int n0 = (tile % nTiles) << 6;
  const int half = lane >> 4, l16 = lane & 15;

  const ushort_t* arow  = A + (size_t)(m0 + l16) * K;
  const ushort_t* brow0 = Bw + (size_t)(n0 + l16) * K;
  const ushort_t* brow1 = brow0 + (size_t)16 * K;
  const ushort_t* brow2 = brow0 + (size_t)32 * K;
  const ushort_t* brow3 = brow0 + (size_t)48 * K;
  const int aoff = half ? 16 : 0;    // A layout: lanes>=16 hold K+16
  const int boff = half ? 16 : 0;    // B layout: lanes>=16 hold K+16

  v8f acc0 = {}, acc1 = {}, acc2 = {}, acc3 = {};
  for (int k0 = 0; k0 < K; k0 += 32) {
    V16 a, b;
    a.u[0] = *(const u32x4*)(arow + k0 + aoff);
    a.u[1] = *(const u32x4*)(arow + k0 + 8 + aoff);
    b.u[0] = *(const u32x4*)(brow0 + k0 + boff);
    b.u[1] = *(const u32x4*)(brow0 + k0 + 8 + boff);
    acc0 = __builtin_amdgcn_wmma_f32_16x16x32_bf16(false, a.v, false, b.v, (short)0, acc0, false, false);
    b.u[0] = *(const u32x4*)(brow1 + k0 + boff);
    b.u[1] = *(const u32x4*)(brow1 + k0 + 8 + boff);
    acc1 = __builtin_amdgcn_wmma_f32_16x16x32_bf16(false, a.v, false, b.v, (short)0, acc1, false, false);
    b.u[0] = *(const u32x4*)(brow2 + k0 + boff);
    b.u[1] = *(const u32x4*)(brow2 + k0 + 8 + boff);
    acc2 = __builtin_amdgcn_wmma_f32_16x16x32_bf16(false, a.v, false, b.v, (short)0, acc2, false, false);
    b.u[0] = *(const u32x4*)(brow3 + k0 + boff);
    b.u[1] = *(const u32x4*)(brow3 + k0 + 8 + boff);
    acc3 = __builtin_amdgcn_wmma_f32_16x16x32_bf16(false, a.v, false, b.v, (short)0, acc3, false, false);
  }
#pragma unroll
  for (int r = 0; r < 8; ++r) {
    const size_t base = (size_t)(m0 + r + half * 8) * N + n0 + l16;
    if (OUT_F32) {
      Cf[base]      = acc0[r];
      Cf[base + 16] = acc1[r];
      Cf[base + 32] = acc2[r];
      Cf[base + 48] = acc3[r];
    } else {
      Cb[base]      = f2bf(acc0[r]);
      Cb[base + 16] = f2bf(acc1[r]);
      Cb[base + 32] = f2bf(acc2[r]);
      Cb[base + 48] = f2bf(acc3[r]);
    }
  }
}

// ------------------------------------------------------------- flash attention
// One wave = one (b,h) x 16 query rows.  Qr/Kr: [B,H,S,64] bf16, Vt: [B,H,64,S] bf16.
// K/V tiles are staged into LDS by the Tensor Data Mover (double-buffered,
// TENSORcnt-tracked) when available; ctx out: [B,S,H*64] bf16.
__global__ __launch_bounds__(32)
void attn_kernel(const ushort_t* __restrict__ Qr, const ushort_t* __restrict__ Kr,
                 const ushort_t* __restrict__ Vt, ushort_t* __restrict__ Ctx, int S) {
#if HAVE_TDM
  __shared__ __align__(16) ushort_t Kst[2][32 * 64];   // [key][d]
  __shared__ __align__(16) ushort_t Vst[2][64 * 32];   // [d][key]
#endif
  __shared__ __align__(16) ushort_t Plds[16 * 32];

  const int nMt = S >> 4;
  int bid = blockIdx.x;
  const int mt = bid % nMt;  bid /= nMt;
  const int h  = bid & 15;
  const int b  = bid >> 4;
  const int m0 = mt << 4;

  const int lane = threadIdx.x;
  const int half = lane >> 4, l16 = lane & 15;
  const size_t bhS = ((size_t)(b * 16 + h)) * S;

  // Q A-tiles (K = d, two chunks of 32), resident for the whole loop.
  const ushort_t* qbase = Qr + (bhS + m0 + l16) * 64;
  const int aoff = half ? 16 : 0;
  V16 aQ0, aQ1;
  aQ0.u[0] = *(const u32x4*)(qbase + aoff);
  aQ0.u[1] = *(const u32x4*)(qbase + 8 + aoff);
  aQ1.u[0] = *(const u32x4*)(qbase + 32 + aoff);
  aQ1.u[1] = *(const u32x4*)(qbase + 40 + aoff);

  float rowM[8], rowL[8];
#pragma unroll
  for (int r = 0; r < 8; ++r) { rowM[r] = -__builtin_inff(); rowL[r] = 0.0f; }
  v8f acc0 = {}, acc1 = {}, acc2 = {}, acc3 = {};

  const int boff = half ? 16 : 0;
  const int jbMax = (m0 + 15) >> 5;

#if HAVE_TDM
  const ushort_t* kTile = Kr + bhS * 64;                 // [S,64], row stride 64
  const ushort_t* vTile = Vt + bhS * 64;                 // [64,S], row stride S
  // Prologue: DMA tiles for jb=0 into buffer 0.
  tdm_load_2d((unsigned)(unsigned long)&Kst[0][0], kTile, 64, (unsigned)S, 64, 32, 64ull);
  tdm_load_2d((unsigned)(unsigned long)&Vst[0][0], vTile, (unsigned)S, 64, 32, 64, (unsigned long long)S);
#endif

  for (int jb = 0; jb <= jbMax; ++jb) {
    const int j0 = jb << 5;
    const int buf = jb & 1;
#if HAVE_TDM
    if (jb < jbMax) {  // prefetch next key block into the other buffer
      const int jn = j0 + 32;
      tdm_load_2d((unsigned)(unsigned long)&Kst[buf ^ 1][0], kTile + (size_t)jn * 64,
                  64, (unsigned)S, 64, 32, 64ull);
      tdm_load_2d((unsigned)(unsigned long)&Vst[buf ^ 1][0], vTile + jn,
                  (unsigned)S, 64, 32, 64, (unsigned long long)S);
      WAIT_TENSOR(2);      // current buffer's pair has completed (in-order)
    } else {
      WAIT_TENSOR(0);
    }
    asm volatile("" ::: "memory");
#endif
    // ---- scores = Q @ K^T (32 keys = two 16-wide N tiles)
    v8f sc0 = {}, sc1 = {};
    {
#if HAVE_TDM
      const ushort_t* kb  = &Kst[buf][(size_t)l16 * 64];
      const ushort_t* kb1 = kb + 16 * 64;
#else
      const ushort_t* kb  = Kr + (bhS + j0 + l16) * 64;
      const ushort_t* kb1 = Kr + (bhS + j0 + 16 + l16) * 64;
#endif
      V16 bk;
      bk.u[0] = *(const u32x4*)(kb + boff);
      bk.u[1] = *(const u32x4*)(kb + 8 + boff);
      sc0 = __builtin_amdgcn_wmma_f32_16x16x32_bf16(false, aQ0.v, false, bk.v, (short)0, sc0, false, false);
      bk.u[0] = *(const u32x4*)(kb + 32 + boff);
      bk.u[1] = *(const u32x4*)(kb + 40 + boff);
      sc0 = __builtin_amdgcn_wmma_f32_16x16x32_bf16(false, aQ1.v, false, bk.v, (short)0, sc0, false, false);
      bk.u[0] = *(const u32x4*)(kb1 + boff);
      bk.u[1] = *(const u32x4*)(kb1 + 8 + boff);
      sc1 = __builtin_amdgcn_wmma_f32_16x16x32_bf16(false, aQ0.v, false, bk.v, (short)0, sc1, false, false);
      bk.u[0] = *(const u32x4*)(kb1 + 32 + boff);
      bk.u[1] = *(const u32x4*)(kb1 + 40 + boff);
      sc1 = __builtin_amdgcn_wmma_f32_16x16x32_bf16(false, aQ1.v, false, bk.v, (short)0, sc1, false, false);
    }
    // ---- scale + causal mask
    const bool needMask = (j0 + 31) > m0;
#pragma unroll
    for (int r = 0; r < 8; ++r) {
      const int row = m0 + r + half * 8;
      float s0 = sc0[r] * 0.125f;
      float s1 = sc1[r] * 0.125f;
      if (needMask) {
        if (j0 + l16 > row)      s0 = -__builtin_inff();
        if (j0 + 16 + l16 > row) s1 = -__builtin_inff();
      }
      sc0[r] = s0; sc1[r] = s1;
    }
    // ---- online softmax (row reductions across the 16 lanes of each half)
#pragma unroll
    for (int r = 0; r < 8; ++r) {
      float v = fmaxf(sc0[r], sc1[r]);
      v = fmaxf(v, __shfl_xor(v, 1, 32));
      v = fmaxf(v, __shfl_xor(v, 2, 32));
      v = fmaxf(v, __shfl_xor(v, 4, 32));
      v = fmaxf(v, __shfl_xor(v, 8, 32));
      const float newM = fmaxf(rowM[r], v);
      const float scl  = __expf(rowM[r] - newM);
      rowM[r] = newM;
      const float p0 = __expf(sc0[r] - newM);
      const float p1 = __expf(sc1[r] - newM);
      sc0[r] = p0; sc1[r] = p1;
      float sum = p0 + p1;
      sum += __shfl_xor(sum, 1, 32);
      sum += __shfl_xor(sum, 2, 32);
      sum += __shfl_xor(sum, 4, 32);
      sum += __shfl_xor(sum, 8, 32);
      rowL[r] = rowL[r] * scl + sum;
      acc0[r] *= scl; acc1[r] *= scl; acc2[r] *= scl; acc3[r] *= scl;
    }
    // ---- P (C-layout) -> LDS row-major -> reload in A-layout
#pragma unroll
    for (int r = 0; r < 8; ++r) {
      const int m = r + half * 8;
      Plds[m * 32 + l16]      = f2bf(sc0[r]);
      Plds[m * 32 + 16 + l16] = f2bf(sc1[r]);
    }
    asm volatile("s_wait_dscnt 0" ::: "memory");
    V16 pA;
    {
      const ushort_t* prow = &Plds[l16 * 32 + (half ? 16 : 0)];
      pA.u[0] = *(const u32x4*)(prow);
      pA.u[1] = *(const u32x4*)(prow + 8);
    }
    asm volatile("s_wait_dscnt 0" ::: "memory");
    // ---- ctx += P @ V
    {
#if HAVE_TDM
      const ushort_t* vb0 = &Vst[buf][(size_t)l16 * 32];
      const size_t vstride = 16 * 32;       // 16 d-rows of 32 keys
#else
      const ushort_t* vb0 = Vt + (bhS * 64 + (size_t)l16 * S) + j0;
      const size_t vstride = (size_t)16 * S;
#endif
      V16 bv;
      bv.u[0] = *(const u32x4*)(vb0 + boff);
      bv.u[1] = *(const u32x4*)(vb0 + 8 + boff);
      acc0 = __builtin_amdgcn_wmma_f32_16x16x32_bf16(false, pA.v, false, bv.v, (short)0, acc0, false, false);
      const ushort_t* vb1 = vb0 + vstride;
      bv.u[0] = *(const u32x4*)(vb1 + boff);
      bv.u[1] = *(const u32x4*)(vb1 + 8 + boff);
      acc1 = __builtin_amdgcn_wmma_f32_16x16x32_bf16(false, pA.v, false, bv.v, (short)0, acc1, false, false);
      const ushort_t* vb2 = vb1 + vstride;
      bv.u[0] = *(const u32x4*)(vb2 + boff);
      bv.u[1] = *(const u32x4*)(vb2 + 8 + boff);
      acc2 = __builtin_amdgcn_wmma_f32_16x16x32_bf16(false, pA.v, false, bv.v, (short)0, acc2, false, false);
      const ushort_t* vb3 = vb2 + vstride;
      bv.u[0] = *(const u32x4*)(vb3 + boff);
      bv.u[1] = *(const u32x4*)(vb3 + 8 + boff);
      acc3 = __builtin_amdgcn_wmma_f32_16x16x32_bf16(false, pA.v, false, bv.v, (short)0, acc3, false, false);
    }
  }
  // ---- epilogue: ctx / L, write bf16 [B,S,H*64]
#pragma unroll
  for (int r = 0; r < 8; ++r) {
    const float invL = 1.0f / rowL[r];
    const size_t base = ((size_t)b * S + (m0 + r + half * 8)) * 1024 + h * 64 + l16;
    Ctx[base]      = f2bf(acc0[r] * invL);
    Ctx[base + 16] = f2bf(acc1[r] * invL);
    Ctx[base + 32] = f2bf(acc2[r] * invL);
    Ctx[base + 48] = f2bf(acc3[r] * invL);
  }
}

// ------------------------------------------------------------------- launcher
extern "C" void kernel_launch(void* const* d_in, const int* in_sizes, int n_in,
                              void* d_out, int out_size, void* d_ws, size_t ws_size,
                              hipStream_t stream) {
  const float* x  = (const float*)d_in[0];
  const float* Wq = (const float*)d_in[1];
  const float* Wk = (const float*)d_in[2];
  const float* Wv = (const float*)d_in[3];
  const float* Wo = (const float*)d_in[4];
  const int*   tp = (const int*)d_in[5];

  const int D = 1024, H = 16;
  const int S = in_sizes[5];
  const int B = in_sizes[0] / (S * D);
  const size_t BS = (size_t)B * S;

  char* w = (char*)d_ws;
  ushort_t* xb  = (ushort_t*)w; w += BS * D * 2;
  ushort_t* wqb = (ushort_t*)w; w += (size_t)D * D * 2;
  ushort_t* wkb = (ushort_t*)w; w += (size_t)D * D * 2;
  ushort_t* wvb = (ushort_t*)w; w += (size_t)D * D * 2;
  ushort_t* wob = (ushort_t*)w; w += (size_t)D * D * 2;
  ushort_t* qb  = (ushort_t*)w; w += BS * D * 2;
  ushort_t* kb  = (ushort_t*)w; w += BS * D * 2;
  ushort_t* vb  = (ushort_t*)w; w += BS * D * 2;
  ushort_t* qr  = (ushort_t*)w; w += BS * D * 2;
  ushort_t* kr  = (ushort_t*)w; w += BS * D * 2;
  ushort_t* vt  = (ushort_t*)w; w += BS * D * 2;
  ushort_t* ctx = (ushort_t*)w; w += BS * D * 2;

  // 1) fp32 -> bf16
  {
    int n = (int)(BS * D);
    cvt_f32_to_bf16<<<(n + 255) / 256, 256, 0, stream>>>(x, xb, n);
    int nw = D * D;
    cvt_f32_to_bf16<<<(nw + 255) / 256, 256, 0, stream>>>(Wq, wqb, nw);
    cvt_f32_to_bf16<<<(nw + 255) / 256, 256, 0, stream>>>(Wk, wkb, nw);
    cvt_f32_to_bf16<<<(nw + 255) / 256, 256, 0, stream>>>(Wv, wvb, nw);
    cvt_f32_to_bf16<<<(nw + 255) / 256, 256, 0, stream>>>(Wo, wob, nw);
  }
  // 2) QKV projections
  {
    const int M = (int)BS, N = D, K = D;
    const int blocks = ((M >> 4) * (N >> 6) + 7) / 8;
    gemm_bf16_nt<false><<<blocks, 256, 0, stream>>>(xb, wqb, nullptr, qb, M, N, K);
    gemm_bf16_nt<false><<<blocks, 256, 0, stream>>>(xb, wkb, nullptr, kb, M, N, K);
    gemm_bf16_nt<false><<<blocks, 256, 0, stream>>>(xb, wvb, nullptr, vb, M, N, K);
  }
  // 3) RoPE relayout Q,K ; transpose V
  {
    int total = (int)(BS * H * 32);
    rope_kernel<<<(total + 255) / 256, 256, 0, stream>>>(qb, kb, tp, qr, kr, S, total);
    int n = (int)(BS * D);
    transpose_v<<<(n + 255) / 256, 256, 0, stream>>>(vb, vt, S, n);
  }
  // 4) flash attention
  {
    const int blocks = B * H * (S >> 4);
    attn_kernel<<<blocks, 32, 0, stream>>>(qr, kr, vt, ctx, S);
  }
  // 5) output projection (fp32 out)
  {
    const int M = (int)BS, N = D, K = D;
    const int blocks = ((M >> 4) * (N >> 6) + 7) / 8;
    gemm_bf16_nt<true><<<blocks, 256, 0, stream>>>(ctx, wob, (float*)d_out, nullptr, M, N, K);
  }
}
